// BahdanauAttention_54485955117307
// MI455X (gfx1250) — matrix operand order repaired and checked by reading, not compiled, not run on hardware
//
#include <hip/hip_runtime.h>

// ---------------------------------------------------------------------------
// Bahdanau attention for MI455X (gfx1250, wave32).
// enc_proj GEMM = 34.4 GFLOP over 268 MB fp32 (~128 FLOP/byte): to stay at the
// 23.3 TB/s HBM roof the math must run ~3 PFLOP/s, so convert fp32 -> bf16
// in-kernel and use V_WMMA_F32_16X16X32_BF16 (f32 accumulate).
// Register-block 4 att tiles per wave: A fragment loaded once per K-step and
// reused 4x -> 2.5 ds_load_b128 per WMMA. tanh epilogue uses the gfx1250
// hardware V_TANH_F32 (TRANS op, co-executes with the XDL WMMA pipe).
// ---------------------------------------------------------------------------

#define B_    64
#define S_    2048
#define E_    512
#define A_    256
#define SROWS 128            // s-rows per block (8 waves x 16 rows)
#define LDA   (E_ + 8)       // padded LDS row length in bf16 elements
#define BCOLS 64             // att columns staged per outer iteration
#define NBT   (A_ / BCOLS)   // 4 outer iterations
#define KSTEP 32

typedef __attribute__((ext_vector_type(8)))  float          v8f;
typedef __attribute__((ext_vector_type(8)))  unsigned short v8us;
typedef __attribute__((ext_vector_type(16))) __bf16         v16bf;

union FragU { v8us h[2]; v16bf bf; };

__device__ __forceinline__ unsigned short f2bf(float f) {
  unsigned u = __builtin_bit_cast(unsigned, f);
  u += 0x7FFFu + ((u >> 16) & 1u);       // round-to-nearest-even
  return (unsigned short)(u >> 16);
}

// Branch-free tanh: prefer gfx1250's hardware V_TANH_F32.
__device__ __forceinline__ float fast_tanh(float x) {
#if __has_builtin(__builtin_amdgcn_tanhf)
  return __builtin_amdgcn_tanhf(x);
#elif __has_builtin(__builtin_amdgcn_tanh_f32)
  return __builtin_amdgcn_tanh_f32(x);
#else
  float xc = fminf(fmaxf(x, -10.f), 10.f);
  float e  = __expf(2.f * xc);
  return (e - 1.f) * __builtin_amdgcn_rcpf(e + 1.f);
#endif
}

// ---------------------------------------------------------------------------
// Kernel 1: fused enc_proj (bf16 WMMA) + dec_proj + tanh + w_att dot + mask.
// grid = (S_/SROWS, B_), block = 256 threads (8 waves).
// Writes raw masked scores into the weights slot of d_out.
// ---------------------------------------------------------------------------
extern "C" __global__ __launch_bounds__(256)
void bahdanau_scores(const float* __restrict__ enc,
                     const float* __restrict__ dec,
                     const int*   __restrict__ lens,
                     const float* __restrict__ Wenc,
                     const float* __restrict__ benc,
                     const float* __restrict__ Wdec,
                     const float* __restrict__ bdec,
                     const float* __restrict__ watt,
                     const float* __restrict__ batt,
                     float* __restrict__ scores)
{
  extern __shared__ char smem[];
  unsigned short* ldsA  = (unsigned short*)smem;          // SROWS x LDA bf16
  unsigned short* ldsB  = ldsA + SROWS * LDA;             // BCOLS x LDA bf16
  float*          dpLds = (float*)(ldsB + BCOLS * LDA);   // A_ floats
  float*          wLds  = dpLds + A_;                     // A_ floats

  const int t  = threadIdx.x;
  const int b  = blockIdx.y;
  const int s0 = blockIdx.x * SROWS;

  // ---- stage A: 128 x 512 fp32 -> bf16 in LDS (coalesced float4 loads) ----
  {
    const float4* encv = (const float4*)(enc + ((size_t)b * S_ + s0) * E_);
    #pragma unroll 4
    for (int i = 0; i < (SROWS * E_ / 4) / 256; ++i) {    // 64 iters
      int f   = t + i * 256;
      int row = f >> 7;                                   // / (E_/4)
      int c4  = f & 127;
      float4 v = encv[(size_t)row * (E_ / 4) + c4];
      unsigned short* dst = &ldsA[row * LDA + (c4 << 2)];
      dst[0] = f2bf(v.x); dst[1] = f2bf(v.y);
      dst[2] = f2bf(v.z); dst[3] = f2bf(v.w);
    }
  }

  // ---- dec_proj + both biases, computed redundantly per block (tiny) ----
  {
    float acc = benc[t] + bdec[t];
    const float* dh = dec + (size_t)b * E_;
    #pragma unroll 8
    for (int k = 0; k < E_; ++k)
      acc += dh[k] * Wdec[(size_t)k * A_ + t];
    dpLds[t] = acc;
    wLds[t]  = watt[t];
  }
  __syncthreads();

  const int wave = t >> 5;
  const int lane = t & 31;
  const int half = lane >> 4;     // which 16-lane half
  const int l16  = lane & 15;
  const unsigned short* aRow = &ldsA[(wave * 16 + l16) * LDA];

  float sacc[8];
  #pragma unroll
  for (int i = 0; i < 8; ++i) sacc[i] = 0.f;

  for (int bt = 0; bt < NBT; ++bt) {
    // ---- stage B tile: Wenc[:, bt*64 .. +64) transposed -> ldsB[n][k] ----
    {
      int n  = t & 63;                                    // 0..63
      int k0 = t >> 6;                                    // 0..3
      const float* src = Wenc + bt * BCOLS + n;
      #pragma unroll 8
      for (int j = 0; j < E_ / 4; ++j) {                  // 128 iters
        int k = k0 + j * 4;
        ldsB[n * LDA + k] = f2bf(src[(size_t)k * A_]);
      }
    }
    __syncthreads();

    // ---- 4 register-blocked 16x16 tiles, K = 512 in steps of 32 ----
    v8f c0 = {0.f,0.f,0.f,0.f,0.f,0.f,0.f,0.f};
    v8f c1 = c0, c2 = c0, c3 = c0;

    #pragma unroll 2
    for (int k0 = 0; k0 < E_; k0 += KSTEP) {
      // A frag: lanes 0-15 hold K k0..k0+15,
      //         lanes 16-31 hold K k0+16..k0+31 (matches B)
      FragU fa;
      int ka = k0 + half * 16;
      fa.h[0] = *(const v8us*)(aRow + ka);
      fa.h[1] = *(const v8us*)(aRow + ka + 8);
      // B frags (same k order as A): lanes 0-15 K=k0..k0+15, lanes 16-31 K=k0+16..k0+31
      int kb = k0 + half * 16;
      FragU fb0, fb1, fb2, fb3;
      const unsigned short* bBase = &ldsB[l16 * LDA + kb];
      fb0.h[0] = *(const v8us*)(bBase);
      fb0.h[1] = *(const v8us*)(bBase + 8);
      fb1.h[0] = *(const v8us*)(bBase + 16 * LDA);
      fb1.h[1] = *(const v8us*)(bBase + 16 * LDA + 8);
      fb2.h[0] = *(const v8us*)(bBase + 32 * LDA);
      fb2.h[1] = *(const v8us*)(bBase + 32 * LDA + 8);
      fb3.h[0] = *(const v8us*)(bBase + 48 * LDA);
      fb3.h[1] = *(const v8us*)(bBase + 48 * LDA + 8);
      c0 = __builtin_amdgcn_wmma_f32_16x16x32_bf16(false, fa.bf, false, fb0.bf,
                                                   (short)0, c0, false, false);
      c1 = __builtin_amdgcn_wmma_f32_16x16x32_bf16(false, fa.bf, false, fb1.bf,
                                                   (short)0, c1, false, false);
      c2 = __builtin_amdgcn_wmma_f32_16x16x32_bf16(false, fa.bf, false, fb2.bf,
                                                   (short)0, c2, false, false);
      c3 = __builtin_amdgcn_wmma_f32_16x16x32_bf16(false, fa.bf, false, fb3.bf,
                                                   (short)0, c3, false, false);
    }

    // ---- epilogue: tanh(enc_proj + dec_proj) * w_att, per lane column ----
    #pragma unroll
    for (int j = 0; j < 4; ++j) {
      int   n   = bt * BCOLS + j * 16 + l16;
      float dpv = dpLds[n];
      float wv  = wLds[n];
      const v8f& c = (j == 0) ? c0 : (j == 1) ? c1 : (j == 2) ? c2 : c3;
      #pragma unroll
      for (int i = 0; i < 8; ++i)         // C VGPR i -> row M = i + 8*half
        sacc[i] += wv * fast_tanh(c[i] + dpv);
    }
    __syncthreads();                       // protect ldsB re-stage
  }

  // ---- reduce over N columns: sum across the 16 lanes of each half ----
  #pragma unroll
  for (int i = 0; i < 8; ++i) {
    float v = sacc[i];
    v += __shfl_xor(v, 1, 32);
    v += __shfl_xor(v, 2, 32);
    v += __shfl_xor(v, 4, 32);
    v += __shfl_xor(v, 8, 32);
    sacc[i] = v;
  }

  const int   len = lens[b];
  const float ba  = batt[0];
  #pragma unroll
  for (int i = 0; i < 8; ++i) {
    if (l16 == i) {
      int   srow = s0 + wave * 16 + half * 8 + i;
      float val  = sacc[i] + ba;
      if (srow >= len) val = -1e9f;
      scores[(size_t)b * S_ + srow] = val;
    }
  }
}

// ---------------------------------------------------------------------------
// Kernel 2: in-place softmax over S per batch row. grid = B_, block = 256.
// ---------------------------------------------------------------------------
extern "C" __global__ __launch_bounds__(256)
void bahdanau_softmax(float* __restrict__ scores)
{
  __shared__ float red[256];
  const int t = threadIdx.x;
  float* row = scores + (size_t)blockIdx.x * S_;

  float v[8];
  float m = -3.4e38f;
  #pragma unroll
  for (int i = 0; i < 8; ++i) { v[i] = row[t + i * 256]; m = fmaxf(m, v[i]); }
  red[t] = m; __syncthreads();
  for (int off = 128; off > 0; off >>= 1) {
    if (t < off) red[t] = fmaxf(red[t], red[t + off]);
    __syncthreads();
  }
  m = red[0]; __syncthreads();

  float s = 0.f;
  #pragma unroll
  for (int i = 0; i < 8; ++i) { v[i] = __expf(v[i] - m); s += v[i]; }
  red[t] = s; __syncthreads();
  for (int off = 128; off > 0; off >>= 1) {
    if (t < off) red[t] += red[t + off];
    __syncthreads();
  }
  const float inv = 1.0f / red[0];
  #pragma unroll
  for (int i = 0; i < 8; ++i) row[t + i * 256] = v[i] * inv;
}

// ---------------------------------------------------------------------------
// Kernel 3: context[b,e] = sum_s weights[b,s] * enc[b,s,e].
// grid = (E_/256, B_), block = 256. Coalesced along E, weights in LDS.
// ---------------------------------------------------------------------------
extern "C" __global__ __launch_bounds__(256)
void bahdanau_context(const float* __restrict__ enc,
                      const float* __restrict__ weights,
                      float* __restrict__ ctx)
{
  __shared__ float wsm[S_];
  const int t = threadIdx.x;
  const int b = blockIdx.y;
  const int e = blockIdx.x * 256 + t;

  #pragma unroll
  for (int i = 0; i < S_ / 256; ++i)
    wsm[t + i * 256] = weights[(size_t)b * S_ + t + i * 256];
  __syncthreads();

  const float* ep = enc + (size_t)b * S_ * E_ + e;
  float acc = 0.f;
  #pragma unroll 8
  for (int s = 0; s < S_; ++s)
    acc += wsm[s] * ep[(size_t)s * E_];
  ctx[(size_t)b * E_ + e] = acc;
}

// ---------------------------------------------------------------------------
// Host-side launcher.
// ---------------------------------------------------------------------------
extern "C" void kernel_launch(void* const* d_in, const int* in_sizes, int n_in,
                              void* d_out, int out_size, void* d_ws, size_t ws_size,
                              hipStream_t stream) {
  (void)in_sizes; (void)n_in; (void)out_size; (void)d_ws; (void)ws_size;
  const float* enc  = (const float*)d_in[0];
  const float* dec  = (const float*)d_in[1];
  const int*   lens = (const int*)d_in[2];
  const float* Wenc = (const float*)d_in[3];
  const float* benc = (const float*)d_in[4];
  const float* Wdec = (const float*)d_in[5];
  const float* bdec = (const float*)d_in[6];
  const float* watt = (const float*)d_in[7];
  const float* batt = (const float*)d_in[8];

  float* out = (float*)d_out;
  float* ctx = out;                 // [B_, E_]   = 32768 floats
  float* wts = out + B_ * E_;       // [B_, S_]   = 131072 floats

  const size_t smemScores =
      (size_t)(SROWS * LDA + BCOLS * LDA) * sizeof(unsigned short) +
      (size_t)(2 * A_) * sizeof(float);   // ~202 KB < 320 KB WGP LDS

  dim3 g1(S_ / SROWS, B_);
  bahdanau_scores<<<g1, 256, smemScores, stream>>>(
      enc, dec, lens, Wenc, benc, Wdec, bdec, watt, batt, wts);

  bahdanau_softmax<<<B_, 256, 0, stream>>>(wts);

  dim3 g3(E_ / 256, B_);
  bahdanau_context<<<g3, 256, 0, stream>>>(enc, wts, ctx);
}
